// CamembertSelfAttention_8839042695305
// MI455X (gfx1250) — matrix operand order repaired and checked by reading, not compiled, not run on hardware
//
#include <hip/hip_runtime.h>
#include <hip/hip_bf16.h>

// ---------------------------------------------------------------------------
// CamembertSelfAttention for MI455X (gfx1250, wave32, WMMA bf16 16x16x32)
// B=4, S=2048, HIDDEN=1024, H=16, Dh=64
// Stage 1: fused QKV GEMM (f32 -> bf16 WMMA, f32 accum), outputs bf16 [B,H,S,Dh]
//          register-staged double-buffered LDS, 6 WMMAs per wave per K-step.
// Stage 2: flash attention (QK^T, online softmax, PV), f32 out [B,S,1024]
//          double-buffered K/V chunks, grouped fragment loads before WMMA.
// ---------------------------------------------------------------------------

#define BATCH   4
#define SEQ     2048
#define HID     1024
#define NHEADS  16
#define HDIM    64
#define ROWS    (BATCH * SEQ)   // 8192

typedef __bf16 bf16_t;
typedef __attribute__((ext_vector_type(16))) __bf16 v16bf;
typedef __attribute__((ext_vector_type(8)))  float  v8f;

union FragAB {
    v16bf v;
    uint4 q[2];
};

__device__ __forceinline__ unsigned short f2bf(float f) {
    union { __bf16 h; unsigned short u; } cv;
    cv.h = (__bf16)f;                      // native f32->bf16 convert
    return cv.u;
}
__device__ __forceinline__ unsigned pack2bf(float lo, float hi) {
    return (unsigned)f2bf(lo) | ((unsigned)f2bf(hi) << 16);
}

__device__ __forceinline__ float rmax16(float v) {
    #pragma unroll
    for (int m = 1; m < 16; m <<= 1) v = fmaxf(v, __shfl_xor(v, m, 32));
    return v;
}
__device__ __forceinline__ float rsum16(float v) {
    #pragma unroll
    for (int m = 1; m < 16; m <<= 1) v += __shfl_xor(v, m, 32);
    return v;
}

// ===========================================================================
// Stage 1: QKV projection.  Grid: (ROWS/64, HID/64), block 256 (8 waves).
// Wave w: m_sub = w&3 (16 rows), n_subs = {2*(w>>2), 2*(w>>2)+1}.
// ===========================================================================
__global__ __launch_bounds__(256)
void qkv_gemm_kernel(const float* __restrict__ X,
                     const float* __restrict__ Wq, const float* __restrict__ bq,
                     const float* __restrict__ Wk, const float* __restrict__ bk,
                     const float* __restrict__ Wv, const float* __restrict__ bv,
                     unsigned short* __restrict__ Qb,
                     unsigned short* __restrict__ Kb,
                     unsigned short* __restrict__ Vb)
{
    __shared__ unsigned short sX[2][64 * 32];     // [m][k], stride 32
    __shared__ unsigned short sW[2][3][64 * 40];  // transposed [n][k], stride 40

    const int tid  = threadIdx.x;
    const int lane = tid & 31;
    const int wave = tid >> 5;
    const int msub = wave & 3;
    const int nsub0 = (wave >> 2) * 2;
    const int halfHi = (lane >= 16);

    const int mbase = blockIdx.x * 64;
    const int nbase = blockIdx.y * 64;

    const float* const Wmat[3] = { Wq, Wk, Wv };

    v8f acc[3][2];
    #pragma unroll
    for (int m = 0; m < 3; ++m)
        #pragma unroll
        for (int j = 0; j < 2; ++j) acc[m][j] = (v8f){0,0,0,0,0,0,0,0};

    // register staging for double buffering
    float2 xr[4];
    float2 wr[3][4];

    auto loadRegs = [&](int kk) {
        #pragma unroll
        for (int it = 0; it < 4; ++it) {
            int idx = tid + it * 256;              // float2 units, 0..1023
            int m = idx >> 4, kp = idx & 15;
            xr[it] = *(const float2*)&X[(size_t)(mbase + m) * HID + kk + 2 * kp];
        }
        #pragma unroll
        for (int mat = 0; mat < 3; ++mat)
            #pragma unroll
            for (int it = 0; it < 4; ++it) {
                int idx = tid + it * 256;
                int krow = idx >> 5, np = idx & 31;
                wr[mat][it] = *(const float2*)&Wmat[mat][(size_t)(kk + krow) * HID + nbase + 2 * np];
            }
    };
    auto storeRegs = [&](int buf) {
        #pragma unroll
        for (int it = 0; it < 4; ++it) {
            int idx = tid + it * 256;
            int m = idx >> 4, kp = idx & 15;
            *(unsigned*)&sX[buf][m * 32 + 2 * kp] = pack2bf(xr[it].x, xr[it].y);
        }
        #pragma unroll
        for (int mat = 0; mat < 3; ++mat)
            #pragma unroll
            for (int it = 0; it < 4; ++it) {
                int idx = tid + it * 256;
                int krow = idx >> 5, np = idx & 31;
                sW[buf][mat][(2 * np) * 40 + krow]     = f2bf(wr[mat][it].x);
                sW[buf][mat][(2 * np + 1) * 40 + krow] = f2bf(wr[mat][it].y);
            }
    };
    auto compute = [&](int buf) {
        // A fragment: 16x32 of X rows [msub*16 .. +15]
        FragAB a;
        {
            int row = msub * 16 + (lane & 15);
            int base = row * 32 + (halfHi ? 8 : 0);
            a.q[0] = *(const uint4*)(&sX[buf][base]);
            a.q[1] = *(const uint4*)(&sX[buf][base + 16]);
        }
        // grouped B-fragment loads, then batched WMMAs
        FragAB bfr[6];
        #pragma unroll
        for (int mat = 0; mat < 3; ++mat)
            #pragma unroll
            for (int j = 0; j < 2; ++j) {
                int ncol = (nsub0 + j) * 16 + (lane & 15);
                int base = ncol * 40 + (halfHi ? 8 : 0);
                bfr[mat * 2 + j].q[0] = *(const uint4*)(&sW[buf][mat][base]);
                bfr[mat * 2 + j].q[1] = *(const uint4*)(&sW[buf][mat][base + 16]);
            }
        #pragma unroll
        for (int mat = 0; mat < 3; ++mat)
            #pragma unroll
            for (int j = 0; j < 2; ++j)
                acc[mat][j] = __builtin_amdgcn_wmma_f32_16x16x32_bf16(
                    false, a.v, false, bfr[mat * 2 + j].v, (short)0, acc[mat][j], false, false);
    };

    const int NSTEP = HID / 32;
    int cur = 0;
    loadRegs(0);
    storeRegs(0);
    __syncthreads();
    for (int step = 0; step < NSTEP; ++step) {
        if (step + 1 < NSTEP) loadRegs((step + 1) * 32);   // prefetch next tile
        compute(cur);
        if (step + 1 < NSTEP) {
            storeRegs(cur ^ 1);
            __syncthreads();
            cur ^= 1;
        }
    }

    // ---- epilogue: bias add, scatter into [B,H,S,Dh] bf16 ----
    const float* const bias[3] = { bq, bk, bv };
    unsigned short* const dst[3] = { Qb, Kb, Vb };
    #pragma unroll
    for (int mat = 0; mat < 3; ++mat) {
        #pragma unroll
        for (int j = 0; j < 2; ++j) {
            int n_g = nbase + (nsub0 + j) * 16 + (lane & 15);
            int h = n_g >> 6, d = n_g & 63;
            float bia = bias[mat][n_g];
            #pragma unroll
            for (int r = 0; r < 8; ++r) {
                int m_g = mbase + msub * 16 + (halfHi ? (r + 8) : r);
                int b = m_g >> 11, s = m_g & 2047;
                dst[mat][(((size_t)(b * NHEADS + h) * SEQ) + s) * HDIM + d] =
                    f2bf(acc[mat][j][r] + bia);
            }
        }
    }
}

// ===========================================================================
// Stage 2: flash attention. Grid: (SEQ/128, B*H), block 256 (8 waves).
// Wave w handles q rows [qbase .. qbase+15], full Dh=64 context.
// ===========================================================================
__global__ __launch_bounds__(256)
void attn_kernel(const unsigned short* __restrict__ Qb,
                 const unsigned short* __restrict__ Kb,
                 const unsigned short* __restrict__ Vb,
                 const float* __restrict__ mask,
                 float* __restrict__ out)
{
    __shared__ unsigned short sK[2][32 * 64];     // [sk][d], stride 64
    __shared__ unsigned short sVt[2][64 * 40];    // transposed [d][sk], stride 40
    __shared__ unsigned short sP[8][16 * 32];     // per-wave probs [row][sk]

    const int tid  = threadIdx.x;
    const int lane = tid & 31;
    const int wave = tid >> 5;
    const int halfHi = (lane >= 16);

    const int bh = blockIdx.y;                    // b*16 + h
    const int b  = bh >> 4;
    const int h  = bh & 15;
    const int qbase = blockIdx.x * 128 + wave * 16;

    // ---- Q fragments for the two Dh k-steps (d 0..31, 32..63) ----
    FragAB aq[2];
    {
        int qrow = qbase + (lane & 15);
        const unsigned short* rp = Qb + ((size_t)bh * SEQ + qrow) * HDIM;
        #pragma unroll
        for (int ks = 0; ks < 2; ++ks) {
            int base = ks * 32 + (halfHi ? 8 : 0);
            aq[ks].q[0] = *(const uint4*)(rp + base);
            aq[ks].q[1] = *(const uint4*)(rp + base + 16);
        }
    }

    v8f ctx[4];
    #pragma unroll
    for (int j = 0; j < 4; ++j) ctx[j] = (v8f){0,0,0,0,0,0,0,0};
    float mrun[8], lrun[8];
    #pragma unroll
    for (int r = 0; r < 8; ++r) { mrun[r] = -1.0e30f; lrun[r] = 0.0f; }

    const float inv_sqrt_d = 0.125f;              // 1/sqrt(64)

    // register staging for K/V double buffering (u32 = 2 bf16)
    unsigned kr[4], vr[4];
    auto loadRegs = [&](int c) {
        const unsigned* Kg = (const unsigned*)(Kb + ((size_t)bh * SEQ + c * 32) * HDIM);
        const unsigned* Vg = (const unsigned*)(Vb + ((size_t)bh * SEQ + c * 32) * HDIM);
        #pragma unroll
        for (int it = 0; it < 4; ++it) {
            int idx = tid + it * 256;              // u32 units, 0..1023
            kr[it] = Kg[idx];
            vr[it] = Vg[idx];
        }
    };
    auto storeRegs = [&](int buf) {
        #pragma unroll
        for (int it = 0; it < 4; ++it) {
            int idx = tid + it * 256;
            int sk = idx >> 5, dp = idx & 31;      // d = 2*dp
            *(unsigned*)&sK[buf][sk * 64 + 2 * dp] = kr[it];
            sVt[buf][(2 * dp) * 40 + sk]     = (unsigned short)(vr[it] & 0xffffu);
            sVt[buf][(2 * dp + 1) * 40 + sk] = (unsigned short)(vr[it] >> 16);
        }
    };

    auto computeChunk = [&](int buf, int c) {
        // ---- grouped B-fragment loads for both score tiles x both k-steps ----
        FragAB bk_[4];                             // [ks*2 + tile]
        #pragma unroll
        for (int ks = 0; ks < 2; ++ks)
            #pragma unroll
            for (int t = 0; t < 2; ++t) {
                int base = ((lane & 15) + t * 16) * 64 + ks * 32 + (halfHi ? 8 : 0);
                bk_[ks * 2 + t].q[0] = *(const uint4*)(&sK[buf][base]);
                bk_[ks * 2 + t].q[1] = *(const uint4*)(&sK[buf][base + 16]);
            }
        v8f s0 = (v8f){0,0,0,0,0,0,0,0};
        v8f s1 = (v8f){0,0,0,0,0,0,0,0};
        s0 = __builtin_amdgcn_wmma_f32_16x16x32_bf16(false, aq[0].v, false, bk_[0].v,
                                                     (short)0, s0, false, false);
        s1 = __builtin_amdgcn_wmma_f32_16x16x32_bf16(false, aq[0].v, false, bk_[1].v,
                                                     (short)0, s1, false, false);
        s0 = __builtin_amdgcn_wmma_f32_16x16x32_bf16(false, aq[1].v, false, bk_[2].v,
                                                     (short)0, s0, false, false);
        s1 = __builtin_amdgcn_wmma_f32_16x16x32_bf16(false, aq[1].v, false, bk_[3].v,
                                                     (short)0, s1, false, false);

        // ---- scale + additive mask ----
        float mv0 = mask[b * SEQ + c * 32 + (lane & 15)];
        float mv1 = mask[b * SEQ + c * 32 + 16 + (lane & 15)];
        #pragma unroll
        for (int r = 0; r < 8; ++r) {
            s0[r] = s0[r] * inv_sqrt_d + mv0;
            s1[r] = s1[r] * inv_sqrt_d + mv1;
        }

        // ---- online softmax (row-wise over 32 keys) ----
        float p0[8], p1[8], scale[8];
        #pragma unroll
        for (int r = 0; r < 8; ++r) {
            float cmax = rmax16(fmaxf(s0[r], s1[r]));
            float mnew = fmaxf(mrun[r], cmax);
            scale[r] = __expf(mrun[r] - mnew);
            p0[r] = __expf(s0[r] - mnew);
            p1[r] = __expf(s1[r] - mnew);
            float csum = rsum16(p0[r] + p1[r]);
            lrun[r] = lrun[r] * scale[r] + csum;
            mrun[r] = mnew;
        }
        #pragma unroll
        for (int j = 0; j < 4; ++j)
            #pragma unroll
            for (int r = 0; r < 8; ++r) ctx[j][r] *= scale[r];

        // ---- C-layout -> A-layout via per-wave LDS (wave-private) ----
        unsigned short* pw = &sP[wave][0];
        #pragma unroll
        for (int r = 0; r < 8; ++r) {
            int row = halfHi ? (r + 8) : r;
            pw[row * 32 + (lane & 15)]      = f2bf(p0[r]);
            pw[row * 32 + 16 + (lane & 15)] = f2bf(p1[r]);
        }
        FragAB ap;
        {
            int base = (lane & 15) * 32 + (halfHi ? 8 : 0);
            ap.q[0] = *(const uint4*)(pw + base);
            ap.q[1] = *(const uint4*)(pw + base + 16);
        }

        // ---- ctx += P x V: grouped V-fragment loads, then 4 WMMAs ----
        FragAB bv_[4];
        #pragma unroll
        for (int j = 0; j < 4; ++j) {
            int base = (j * 16 + (lane & 15)) * 40 + (halfHi ? 8 : 0);
            bv_[j].q[0] = *(const uint4*)(&sVt[buf][base]);
            bv_[j].q[1] = *(const uint4*)(&sVt[buf][base + 16]);
        }
        #pragma unroll
        for (int j = 0; j < 4; ++j)
            ctx[j] = __builtin_amdgcn_wmma_f32_16x16x32_bf16(false, ap.v, false, bv_[j].v,
                                                             (short)0, ctx[j], false, false);
    };

    const int NCHUNK = SEQ / 32;
    int cur = 0;
    loadRegs(0);
    storeRegs(0);
    __syncthreads();
    for (int c = 0; c < NCHUNK; ++c) {
        if (c + 1 < NCHUNK) loadRegs(c + 1);       // prefetch next K/V chunk
        computeChunk(cur, c);
        if (c + 1 < NCHUNK) {
            storeRegs(cur ^ 1);
            __syncthreads();
            cur ^= 1;
        }
    }

    // ---- normalize and write [B,S,H*Dh] fp32 ----
    #pragma unroll
    for (int j = 0; j < 4; ++j) {
        int d = j * 16 + (lane & 15);
        #pragma unroll
        for (int r = 0; r < 8; ++r) {
            int qrow = qbase + (halfHi ? (r + 8) : r);
            out[((size_t)(b * SEQ + qrow)) * HID + h * HDIM + d] = ctx[j][r] / lrun[r];
        }
    }
}

// ===========================================================================
extern "C" void kernel_launch(void* const* d_in, const int* in_sizes, int n_in,
                              void* d_out, int out_size, void* d_ws, size_t ws_size,
                              hipStream_t stream) {
    const float* X    = (const float*)d_in[0];
    const float* mask = (const float*)d_in[1];
    const float* Wq   = (const float*)d_in[2];
    const float* bq   = (const float*)d_in[3];
    const float* Wk   = (const float*)d_in[4];
    const float* bk   = (const float*)d_in[5];
    const float* Wv   = (const float*)d_in[6];
    const float* bv   = (const float*)d_in[7];
    float* out = (float*)d_out;

    // workspace: Q,K,V bf16 in [B,H,S,Dh] (16 MB each)
    unsigned short* Qb = (unsigned short*)d_ws;
    unsigned short* Kb = Qb + (size_t)ROWS * HID;
    unsigned short* Vb = Kb + (size_t)ROWS * HID;

    dim3 g1(ROWS / 64, HID / 64);                 // 128 x 16
    qkv_gemm_kernel<<<g1, 256, 0, stream>>>(X, Wq, bq, Wk, bk, Wv, bv, Qb, Kb, Vb);

    dim3 g2(SEQ / 128, BATCH * NHEADS);           // 16 x 64
    attn_kernel<<<g2, 256, 0, stream>>>(Qb, Kb, Vb, mask, out);
}
